// THGCAgent_40346922778758
// MI455X (gfx1250) — matrix operand order repaired and checked by reading, not compiled, not run on hardware
//
#include <hip/hip_runtime.h>

// ---------------------------------------------------------------------------
// 3-layer GCN for MI455X (gfx1250, wave32, WMMA).
//   B=16, N=2048, F=64.  An is built once in bf16 (128 MB -> fits 192MB L2),
//   all GEMMs use v_wmma_f32_16x16x32_bf16 with f32 accumulation.
//   Aggregation stages the shared HW panel in LDS via async-to-LDS copies
//   (double buffered), and software-pipelines the A-tile stream.
// ---------------------------------------------------------------------------

#define B_ 16
#define N_ 2048
#define F_ 64

// 32 rows x 64 halfs, padded to 72 halfs/row (144B) -> conflict-free ds_load_b128
#define PROW 72

typedef __attribute__((ext_vector_type(16))) __bf16          v16bf;
typedef __attribute__((ext_vector_type(8)))  float           v8f;
typedef __attribute__((ext_vector_type(4)))  float           v4f;
typedef __attribute__((ext_vector_type(8)))  unsigned short  v8us;
typedef __attribute__((ext_vector_type(4)))  unsigned short  v4us;

union ABv { v16bf v; v4f q[2]; };

#if defined(__gfx1250__) && __has_builtin(__builtin_amdgcn_global_load_async_to_lds_b128)
#define HAVE_ASYNC 1
#else
#define HAVE_ASYNC 0
#endif

#if HAVE_ASYNC
// Builtin signature (from hipcc diagnostic): param0 = AS1 int4*, param1 = AS3
// int4*, then imm offset + imm cpol.
typedef int v4i_vs __attribute__((vector_size(16)));
typedef __attribute__((address_space(1))) v4i_vs* as1_v4i;
typedef __attribute__((address_space(3))) v4i_vs* as3_v4i;
#if __has_builtin(__builtin_amdgcn_s_wait_asynccnt)
#define WAIT_ASYNC() __builtin_amdgcn_s_wait_asynccnt(0)
#else
#define WAIT_ASYNC() asm volatile("s_wait_asynccnt 0" ::: "memory")
#endif
#else
#define WAIT_ASYNC()
#endif

__device__ __forceinline__ unsigned short f32_to_bf16(float f) {
  unsigned int u = __float_as_uint(f);
  u += 0x7FFFu + ((u >> 16) & 1u);          // round-to-nearest-even
  return (unsigned short)(u >> 16);
}

__device__ __forceinline__ v8f wmma_bf16(v16bf a, v16bf b, v8f c) {
  // (neg_a, A, neg_b, B, c_mod, C, reuse_a, reuse_b)
  return __builtin_amdgcn_wmma_f32_16x16x32_bf16(false, a, false, b,
                                                 (short)0, c, false, false);
}

// ---------------------------------------------------------------------------
// Pass 1: one streaming read of the 256MB f32 adjacency.
// Produces dinv[b,i] = rsqrt(deg) and an 8MB nonzero bitmask so the f32
// adjacency never has to be read again.  One wave per row; lane l covers
// column it*32+l so the or-reduced lane mask is exactly the 32-bit bitmask.
// ---------------------------------------------------------------------------
__global__ void gcn_degmask(const float* __restrict__ adj,
                            float* __restrict__ dinv,
                            unsigned int* __restrict__ mask) {
  const int lane = threadIdx.x & 31;
  const int wave = threadIdx.x >> 5;
  const int row  = blockIdx.x * 8 + wave;            // [0, B*N)
  const float* __restrict__ a = adj + (size_t)row * N_;

  int cnt = 0;
  for (int it = 0; it < N_ / 32; ++it) {
    float v = a[it * 32 + lane];                     // 128B coalesced per wave
    unsigned int m = (v != 0.0f) ? (1u << lane) : 0u;
    #pragma unroll
    for (int off = 16; off; off >>= 1)
      m |= (unsigned int)__shfl_xor((int)m, off, 32);
    cnt += __popc(m);
    if (lane == 0) mask[(size_t)row * (N_ / 32) + it] = m;
  }
  const int i = row & (N_ - 1);
  const float diag = a[i];                           // reference zeroes diag
  const float deg  = (float)cnt + (diag != 0.0f ? 0.0f : 1.0f); // A+I self loop
  if (lane == 0) dinv[row] = rsqrtf(deg);
}

// ---------------------------------------------------------------------------
// Pass 2: An[b,i,j] = ((A|I) ? dinv_i*dinv_j : 0) in bf16, from bitmask only.
// Each thread writes 8 packed bf16 (one 16B store).
// ---------------------------------------------------------------------------
__global__ void gcn_normalize(const unsigned int* __restrict__ mask,
                              const float* __restrict__ dinv,
                              unsigned short* __restrict__ An) {
  const size_t gid = (size_t)blockIdx.x * blockDim.x + threadIdx.x;
  const int    jb  = (int)(gid & (N_ / 8 - 1));      // 0..255
  const size_t row = gid >> 8;                       // b*N + i
  const int    j0  = jb * 8;
  const int    i   = (int)(row & (N_ - 1));
  const size_t bb  = row & ~(size_t)(N_ - 1);        // b*N

  const float di  = dinv[row];
  const v4f  djA  = *(const v4f*)(dinv + bb + j0);
  const v4f  djB  = *(const v4f*)(dinv + bb + j0 + 4);
  const unsigned int mb =
      (mask[row * (N_ / 32) + (j0 >> 5)] >> (j0 & 31)) & 0xFFu;

  v8us r;
  #pragma unroll
  for (int c = 0; c < 8; ++c) {
    const int  j  = j0 + c;
    const bool on = ((mb >> c) & 1u) || (j == i);
    const float dj = (c < 4) ? djA[c] : djB[c - 4];
    r[c] = f32_to_bf16(on ? di * dj : 0.0f);
  }
  *(v8us*)(An + row * N_ + j0) = r;
}

// ---------------------------------------------------------------------------
// f32 -> bf16 converters (x once; the three 64x64 weights once).
// ---------------------------------------------------------------------------
__global__ void gcn_cvt_x(const float* __restrict__ x,
                          unsigned short* __restrict__ h) {
  const size_t idx = ((size_t)blockIdx.x * blockDim.x + threadIdx.x) * 4;
  const v4f v = *(const v4f*)(x + idx);
  v4us r;
  #pragma unroll
  for (int c = 0; c < 4; ++c) r[c] = f32_to_bf16(v[c]);
  *(v4us*)(h + idx) = r;
}

__global__ void gcn_cvt_w(const float* __restrict__ w1,
                          const float* __restrict__ w2,
                          const float* __restrict__ w3,
                          unsigned short* __restrict__ wb) {
  const int idx = blockIdx.x * blockDim.x + threadIdx.x;      // < 3*4096
  const float* src = (idx < 4096) ? w1 : (idx < 8192 ? w2 : w3);
  wb[idx] = f32_to_bf16(src[idx & 4095]);
}

// ---------------------------------------------------------------------------
// HW = H @ W  (M = B*N = 32768, K = 64, Nf = 64), bf16 WMMA, bf16 out.
// Wave owns a 16x64 strip (4 accumulators); 8 waves per block = 128 rows.
// A-tile: lane l<16 -> row l, K+[0..7]; lane>=16 -> row l-16, K+[8..15]
//         (two b128 loads per lane, per 7.12.2 16-bit A layout).
// B-tile: lane%16 = N col; 16 K values per lane in A's K order (strided loads).
// ---------------------------------------------------------------------------
__global__ void gcn_feature(const unsigned short* __restrict__ H,
                            const unsigned short* __restrict__ W,
                            unsigned short* __restrict__ HW) {
  const int lane = threadIdx.x & 31;
  const int wave = threadIdx.x >> 5;
  const int lh = lane & 15, hi = lane >> 4;
  const int i0 = blockIdx.x * 128 + wave * 16;       // [0, B*N)

  const unsigned short* __restrict__ arow = H + (size_t)(i0 + lh) * F_ + hi * 8;
  const unsigned short* __restrict__ brow = W + (size_t)(hi * 8) * F_ + lh;

  v8f acc[4] = {};
  #pragma unroll
  for (int k0 = 0; k0 < F_; k0 += 32) {
    ABv a;
    a.q[0] = *(const v4f*)(arow + k0);
    a.q[1] = *(const v4f*)(arow + k0 + 16);
    const unsigned short* bp = brow + (size_t)k0 * F_;
    #pragma unroll
    for (int t = 0; t < 4; ++t) {
      ABv bm;
      #pragma unroll
      for (int s = 0; s < 16; ++s)
        bm.v[s] = __builtin_bit_cast(__bf16, bp[((s & 7) + (s >> 3) * 16) * F_ + t * 16]);
      acc[t] = wmma_bf16(a.v, bm.v, acc[t]);
    }
  }
  #pragma unroll
  for (int t = 0; t < 4; ++t) {
    const int f = t * 16 + lh;
    #pragma unroll
    for (int r = 0; r < 8; ++r) {
      const int m = i0 + (hi ? r + 8 : r);           // C layout, 7.12.2
      HW[(size_t)m * F_ + f] = f32_to_bf16(acc[t][r]);
    }
  }
}

// ---------------------------------------------------------------------------
// OUT = An @ HW + bias (+ ELU, bf16 out) per layer; last layer f32 -> d_out.
// K = 2048 (64 wmma steps).
//   * HW panel (32 rows x 64 feats, shared by ALL 8 waves of the block) is
//     staged in LDS with global_load_async_to_lds_b128, double-buffered:
//     wait_asynccnt -> barrier -> issue(next) -> compute(cur) -> barrier.
//   * A tiles are register double-buffered so the wait before each WMMA only
//     covers the *next* iteration's loads (partial loadcnt), not everything.
// ---------------------------------------------------------------------------
template <bool LAST>
__global__ void gcn_aggregate(const unsigned short* __restrict__ An,
                              const unsigned short* __restrict__ HW,
                              const float* __restrict__ bias,
                              unsigned short* __restrict__ out_bf,
                              float* __restrict__ out_f32) {
  __shared__ alignas(16) unsigned short panel[2][32 * PROW];

  const int tid  = threadIdx.x;
  const int lane = tid & 31;
  const int wave = tid >> 5;
  const int lh = lane & 15, hi = lane >> 4;
  const int b  = blockIdx.y;
  const int i0 = blockIdx.x * 128 + wave * 16;

  const unsigned short* __restrict__ HWb = HW + (size_t)b * N_ * F_;
  const unsigned short* __restrict__ arow =
      An + ((size_t)b * N_ + (size_t)(i0 + lh)) * N_ + hi * 8;

  // Cooperative panel fill: 256 threads x 16B = 4KB = one 32x64 bf16 panel.
  const int pr = tid >> 3;            // panel row   0..31
  const int pc = (tid & 7) * 8;       // panel col   0,8,..,56 (halfs)
  auto issue_panel = [&](int k0, int buf) {
    const unsigned short* src = HWb + (size_t)(k0 + pr) * F_ + pc;
    unsigned short*       dst = &panel[buf][pr * PROW + pc];
#if HAVE_ASYNC
    __builtin_amdgcn_global_load_async_to_lds_b128((as1_v4i)src, (as3_v4i)dst,
                                                   0, 0);
#else
    *(v4f*)dst = *(const v4f*)src;    // global_load_b128 + ds_store_b128
#endif
  };

  issue_panel(0, 0);

  ABv a_cur;                          // A tile for k-step 0
  a_cur.q[0] = *(const v4f*)(arow);
  a_cur.q[1] = *(const v4f*)(arow + 16);

  v8f acc[4] = {};
  for (int it = 0; it < N_ / 32; ++it) {
    const int buf = it & 1;
    WAIT_ASYNC();                     // this wave's panel[buf] copy landed
    __syncthreads();                  // every wave's copy landed

    if (it + 1 < N_ / 32) issue_panel((it + 1) * 32, buf ^ 1);

    ABv a_nxt = a_cur;                // overlap next A load with WMMAs below
    if (it + 1 < N_ / 32) {
      const unsigned short* ap = arow + (it + 1) * 32;
      __builtin_prefetch(ap + 8 * 32, 0, 0);   // global_prefetch, speculative
      a_nxt.q[0] = *(const v4f*)(ap);
      a_nxt.q[1] = *(const v4f*)(ap + 16);
    }

    const unsigned short* bp = &panel[buf][(hi * 8) * PROW + lh];
    #pragma unroll
    for (int t = 0; t < 4; ++t) {
      ABv bm;                         // column lh, 16 K values in A's K order
      #pragma unroll
      for (int s = 0; s < 16; ++s)
        bm.v[s] = __builtin_bit_cast(__bf16, bp[((s & 7) + (s >> 3) * 16) * PROW + t * 16]);
      acc[t] = wmma_bf16(a_cur.v, bm.v, acc[t]);
    }
    a_cur = a_nxt;
    __syncthreads();                  // all waves done reading panel[buf]
  }

  #pragma unroll
  for (int t = 0; t < 4; ++t) {
    const int f  = t * 16 + lh;
    const float bv = bias[f];
    #pragma unroll
    for (int r = 0; r < 8; ++r) {
      const int m = i0 + (hi ? r + 8 : r);
      float v = acc[t][r] + bv;
      const size_t o = ((size_t)b * N_ + m) * F_ + f;
      if (LAST) {
        out_f32[o] = v;
      } else {
        v = (v > 0.0f) ? v : (__expf(v) - 1.0f);     // ELU, alpha=1
        out_bf[o] = f32_to_bf16(v);
      }
    }
  }
}

// ---------------------------------------------------------------------------
// Launcher.  d_in: x, adj, W1, b1, W2, b2, W3, b3  (all f32).  d_out: f32.
// ws layout (≈151 MB): dinv | mask | An(bf16) | H(bf16) | HW(bf16) | W(bf16)
// ---------------------------------------------------------------------------
extern "C" void kernel_launch(void* const* d_in, const int* in_sizes, int n_in,
                              void* d_out, int out_size, void* d_ws,
                              size_t ws_size, hipStream_t stream) {
  (void)in_sizes; (void)n_in; (void)out_size; (void)ws_size;
  const float* x   = (const float*)d_in[0];
  const float* adj = (const float*)d_in[1];
  const float* W1  = (const float*)d_in[2];
  const float* b1  = (const float*)d_in[3];
  const float* W2  = (const float*)d_in[4];
  const float* b2  = (const float*)d_in[5];
  const float* W3  = (const float*)d_in[6];
  const float* b3  = (const float*)d_in[7];
  float* out = (float*)d_out;

  char*  ws  = (char*)d_ws;
  size_t off = 0;
  auto take = [&](size_t bytes) -> char* {
    char* p = ws + off;
    off += (bytes + 255) & ~(size_t)255;
    return p;
  };
  float*          dinv = (float*)take((size_t)B_ * N_ * 4);
  unsigned int*   mask = (unsigned int*)take((size_t)B_ * N_ * N_ / 8);
  unsigned short* An   = (unsigned short*)take((size_t)B_ * N_ * N_ * 2);
  unsigned short* Hbf  = (unsigned short*)take((size_t)B_ * N_ * F_ * 2);
  unsigned short* HWbf = (unsigned short*)take((size_t)B_ * N_ * F_ * 2);
  unsigned short* Wbf  = (unsigned short*)take((size_t)3 * F_ * F_ * 2);

  gcn_cvt_x  <<<(B_ * N_ * F_ / 4) / 256, 256, 0, stream>>>(x, Hbf);
  gcn_cvt_w  <<<(3 * F_ * F_) / 256,      256, 0, stream>>>(W1, W2, W3, Wbf);
  gcn_degmask<<<(B_ * N_) / 8,            256, 0, stream>>>(adj, dinv, mask);
  gcn_normalize<<<(int)(((size_t)B_ * N_ * N_ / 8) / 256), 256, 0, stream>>>(
      mask, dinv, An);

  const dim3 agg_grid(N_ / 128, B_);
  // layer 1
  gcn_feature<<<(B_ * N_) / 128, 256, 0, stream>>>(Hbf, Wbf + 0 * F_ * F_, HWbf);
  gcn_aggregate<false><<<agg_grid, 256, 0, stream>>>(An, HWbf, b1, Hbf, nullptr);
  // layer 2
  gcn_feature<<<(B_ * N_) / 128, 256, 0, stream>>>(Hbf, Wbf + 1 * F_ * F_, HWbf);
  gcn_aggregate<false><<<agg_grid, 256, 0, stream>>>(An, HWbf, b2, Hbf, nullptr);
  // layer 3 (no ELU, f32 output)
  gcn_feature<<<(B_ * N_) / 128, 256, 0, stream>>>(Hbf, Wbf + 2 * F_ * F_, HWbf);
  gcn_aggregate<true><<<agg_grid, 256, 0, stream>>>(An, HWbf, b3, nullptr, out);
}
